// MultiHeadAttention_60928406061424
// MI455X (gfx1250) — matrix operand order repaired and checked by reading, not compiled, not run on hardware
//
#include <hip/hip_runtime.h>
#include <hip/hip_bf16.h>
#include <math.h>

// ---------- types ----------
typedef __bf16        v16bf __attribute__((ext_vector_type(16)));
typedef float         v8f   __attribute__((ext_vector_type(8)));
typedef unsigned int  v4u   __attribute__((ext_vector_type(4)));

union FragBF {            // 16 bf16 = 32 bytes = two b128 chunks
    v16bf bf;
    v4u   u[2];
};

// f32 -> bf16 round-to-nearest-even
__device__ __forceinline__ unsigned short f2bf(float f) {
    unsigned int u = __float_as_uint(f);
    unsigned int r = (u + 0x7FFFu + ((u >> 16) & 1u)) >> 16;
    return (unsigned short)r;
}

// LDS byte address of a __shared__ object (flat ptr low 32 bits == LDS offset)
#define LDS_ADDR(p) ((unsigned)(size_t)(p))

// async global->LDS b128 copy (GVS mode), tracked by ASYNCcnt
__device__ __forceinline__ void async_b128(unsigned lds, unsigned goff,
                                           unsigned long long sbase) {
    asm volatile("global_load_async_to_lds_b128 %0, %1, %2"
                 :: "v"(lds), "v"(goff), "s"(sbase) : "memory");
}
__device__ __forceinline__ void wait_async0() {
    asm volatile("s_wait_asynccnt 0x0" ::: "memory");
}

#define D_MODEL 1024
#define N_HEADS 16
#define D_HEAD  64
#define SEQ     2048
#define BATCH   4

// ============================================================
// Weight transpose + bf16 convert: WT[n][k] = bf16(W[k][n])
// ============================================================
__global__ __launch_bounds__(256) void transpose_w(const float* __restrict__ W,
                                                   unsigned short* __restrict__ WT) {
    int n = blockIdx.x;                       // output row (N index), 0..1023
    for (int k = threadIdx.x; k < D_MODEL; k += 256)
        WT[(size_t)n * D_MODEL + k] = f2bf(W[(size_t)k * D_MODEL + n]);
}

// ============================================================
// Tiled WMMA GEMM:  C[M,N] = A[M,K] @ WT^T + bias
//   M = 8192, N = K = 1024
//   block tile 128x256, k-step 32, double-buffered LDS, k-loop
//   unrolled x2 so buffer selects are compile-time.
//   8 waves, each computes a 64x64 wave tile (16 WMMA / 16 b128
//   LDS loads per k-step).
//   AMODE 0: A is f32 (VGPR convert path)   AMODE 1: A is bf16 (async DMA path)
//   OMODE 0: bf16 out [B,H,S,Dh]            OMODE 1: f32 out row-major
// ============================================================
template <int AMODE, int OMODE>
__global__ __launch_bounds__(256) void gemm_wmma(const void* __restrict__ Ain,
                                                 const unsigned short* __restrict__ WT,
                                                 const float* __restrict__ bias,
                                                 void* __restrict__ Out) {
    __shared__ unsigned short As[2][128 * 40];   // 128 rows x 32 bf16 (+8 pad), 80B rows
    __shared__ unsigned short Bs[2][256 * 40];   // 256 n-rows x 32 bf16 (transposed weights)

    const int tid  = threadIdx.x;
    const int lane = tid & 31;
    const int w    = tid >> 5;       // wave 0..7
    const int wm   = w >> 2;         // 0..1  -> 64 rows
    const int wn   = w & 3;          // 0..3  -> 64 cols
    const int bm   = blockIdx.x * 128;
    const int bn   = blockIdx.y * 256;

    const int arow = tid >> 1, ahalf = tid & 1;  // A staging: 16 elems/thread
    const int brow = tid;                        // B staging: 32 bf16 (64B)/thread

    // ---- staging helpers ----
    auto issueB = [&](int k0, int buf) {         // async DMA: WT tile, 4x b128/thread
        unsigned lds0 = LDS_ADDR(&Bs[buf][brow * 40]);
        unsigned goff = (unsigned)(((size_t)(bn + brow) * D_MODEL + k0) * 2);
        unsigned long long sb = (unsigned long long)WT;
#pragma unroll
        for (int c = 0; c < 4; ++c) async_b128(lds0 + c * 16, goff + c * 16, sb);
    };
    auto issueA_bf16 = [&](int k0, int buf) {    // async DMA: bf16 A tile, 2x b128/thread
        unsigned lds0 = LDS_ADDR(&As[buf][arow * 40 + ahalf * 16]);
        unsigned goff = (unsigned)(((size_t)(bm + arow) * D_MODEL + k0 + ahalf * 16) * 2);
        unsigned long long sb = (unsigned long long)Ain;
#pragma unroll
        for (int c = 0; c < 2; ++c) async_b128(lds0 + c * 16, goff + c * 16, sb);
    };
    auto loadA_f32 = [&](int k0, float* r) {     // f32 A tile -> regs
        const float4* s4 = (const float4*)((const float*)Ain +
                           (size_t)(bm + arow) * D_MODEL + k0 + ahalf * 16);
#pragma unroll
        for (int c = 0; c < 4; ++c) {
            float4 f = s4[c];
            r[c * 4 + 0] = f.x; r[c * 4 + 1] = f.y; r[c * 4 + 2] = f.z; r[c * 4 + 3] = f.w;
        }
    };
    auto storeA_f32 = [&](int buf, const float* r) {   // convert + LDS store
        union { unsigned short s[16]; v4u u[2]; } pk;
#pragma unroll
        for (int j = 0; j < 16; ++j) pk.s[j] = f2bf(r[j]);
        v4u* dst = (v4u*)&As[buf][arow * 40 + ahalf * 16];
        dst[0] = pk.u[0];
        dst[1] = pk.u[1];
    };

    v8f acc[4][4];
#pragma unroll
    for (int mt = 0; mt < 4; ++mt)
#pragma unroll
        for (int nt = 0; nt < 4; ++nt)
#pragma unroll
            for (int i = 0; i < 8; ++i) acc[mt][nt][i] = 0.0f;

    // ---- prologue: stage k-step 0 into buffer 0 ----
    float regsA[16];
    if (AMODE == 0) { loadA_f32(0, regsA); storeA_f32(0, regsA); }
    else            { issueA_bf16(0, 0); }
    issueB(0, 0);
    wait_async0();
    __syncthreads();

    const int kcA = (lane < 16) ? 0 : 8;
    const int kcB = (lane < 16) ? 0 : 8;

#pragma unroll 2
    for (int step = 0; step < 32; ++step) {
        const int  cur = step & 1;
        const int  k0n = (step + 1) * 32;
        const bool nxt = (step + 1) < 32;

        // issue next tile's staging first so HBM latency hides under WMMA
        if (nxt) {
            issueB(k0n, cur ^ 1);
            if (AMODE == 0) loadA_f32(k0n, regsA);
            else            issueA_bf16(k0n, cur ^ 1);
        }

        // ---- compute 64x64 wave tile: 16 WMMA ----
        FragBF a[4];
#pragma unroll
        for (int mt = 0; mt < 4; ++mt) {
            int r = wm * 64 + mt * 16 + (lane & 15);
            a[mt].u[0] = *(const v4u*)&As[cur][r * 40 + kcA];
            a[mt].u[1] = *(const v4u*)&As[cur][r * 40 + kcA + 16];
        }
#pragma unroll
        for (int nt = 0; nt < 4; ++nt) {
            FragBF bfrag;
            int n = wn * 64 + nt * 16 + (lane & 15);
            bfrag.u[0] = *(const v4u*)&Bs[cur][n * 40 + kcB];
            bfrag.u[1] = *(const v4u*)&Bs[cur][n * 40 + kcB + 16];
#pragma unroll
            for (int mt = 0; mt < 4; ++mt)
                acc[mt][nt] = __builtin_amdgcn_wmma_f32_16x16x32_bf16(
                    false, a[mt].bf, false, bfrag.bf, (short)0, acc[mt][nt], false, false);
        }

        if (nxt && AMODE == 0) storeA_f32(cur ^ 1, regsA);
        wait_async0();
        __syncthreads();
    }

    // ---- epilogue: bias + store ----
#pragma unroll
    for (int mt = 0; mt < 4; ++mt)
#pragma unroll
        for (int nt = 0; nt < 4; ++nt)
#pragma unroll
            for (int i = 0; i < 8; ++i) {
                int row = bm + wm * 64 + mt * 16 + i + ((lane >= 16) ? 8 : 0);
                int col = bn + wn * 64 + nt * 16 + (lane & 15);
                float v = acc[mt][nt][i] + bias[col];
                if (OMODE == 0) {
                    int bb = row >> 11, s = row & 2047, h = col >> 6, dh = col & 63;
                    ((unsigned short*)Out)[(((size_t)(bb * N_HEADS + h) * SEQ + s) << 6) + dh] =
                        f2bf(v);
                } else {
                    ((float*)Out)[(size_t)row * D_MODEL + col] = v;
                }
            }
}

// ============================================================
// Causal flash attention over bf16 Q/K/V in [B,H,S,Dh].
// Block = (q-tile of 128 rows) x (b,h). 8 waves, 16 q-rows each.
// kv tiles of 64, double-buffered: K via async global->LDS DMA,
// V via registers, both staged for tile kt+1 while computing kt.
// Q staging buffer is reused as the P-transpose buffer.
// ============================================================
__global__ __launch_bounds__(256) void flash_attn(const unsigned short* __restrict__ Q,
                                                  const unsigned short* __restrict__ K,
                                                  const unsigned short* __restrict__ V,
                                                  const int* __restrict__ mask,
                                                  unsigned short* __restrict__ Ctx) {
    __shared__ unsigned short QsPt[128 * 72];   // Q tile, then P tile (wave-private rows)
    __shared__ unsigned short Ks[2][64 * 72];   // [kv][dh]
    __shared__ unsigned short Vt[2][64 * 72];   // [dh][kv]  (transposed for AV B-frags)

    const int tid  = threadIdx.x;
    const int lane = tid & 31;
    const int w    = tid >> 5;
    const int qt   = blockIdx.x;          // 0..15
    const int bh   = blockIdx.y;          // 0..63
    const int b    = bh >> 4;
    const int h    = bh & 15;
    const size_t base = (size_t)bh * SEQ * D_HEAD;
    const int q0 = qt * 128;

    // ---- stage Q tile once ----
    {
        int row = tid >> 1, half = tid & 1;
        const v4u* src = (const v4u*)(Q + base + (size_t)(q0 + row) * D_HEAD + half * 32);
        v4u* dst = (v4u*)&QsPt[row * 72 + half * 32];
        dst[0] = src[0]; dst[1] = src[1]; dst[2] = src[2]; dst[3] = src[3];
    }
    __syncthreads();

    // per-wave Q fragments (A-matrix layout), 2 k-steps over Dh=64
    FragBF qf[2];
    {
        int r  = w * 16 + (lane & 15);
        int kc = (lane < 16) ? 0 : 8;
#pragma unroll
        for (int ks = 0; ks < 2; ++ks) {
            qf[ks].u[0] = *(const v4u*)&QsPt[r * 72 + ks * 32 + kc];
            qf[ks].u[1] = *(const v4u*)&QsPt[r * 72 + ks * 32 + kc + 16];
        }
    }

    // ---- staging helpers (kv tiles) ----
    auto stageK = [&](int kv0, int buf) {       // async DMA, one b128 per thread
        int row = tid >> 2, qd = tid & 3;
        unsigned lds0 = LDS_ADDR(&Ks[buf][row * 72 + qd * 8]);
        unsigned goff = (unsigned)((base + (size_t)(kv0 + row) * D_HEAD + qd * 8) * 2);
        async_b128(lds0, goff, (unsigned long long)K);
    };
    auto loadV = [&](int kv0, unsigned short* r) {   // 16 bf16 per thread -> regs
        int row = tid >> 2, c0 = (tid & 3) * 16;
        const v4u* src = (const v4u*)(V + base + (size_t)(kv0 + row) * D_HEAD + c0);
        union { unsigned short s[16]; v4u u[2]; } pk;
        pk.u[0] = src[0]; pk.u[1] = src[1];
#pragma unroll
        for (int j = 0; j < 16; ++j) r[j] = pk.s[j];
    };
    auto storeV = [&](int buf, const unsigned short* r) {  // transposed LDS store
        int row = tid >> 2, c0 = (tid & 3) * 16;
#pragma unroll
        for (int j = 0; j < 16; ++j) Vt[buf][(c0 + j) * 72 + row] = r[j];
    };

    v8f oacc[4];
    float mrow[8], lrow[8];
#pragma unroll
    for (int nt = 0; nt < 4; ++nt)
#pragma unroll
        for (int i = 0; i < 8; ++i) oacc[nt][i] = 0.0f;
#pragma unroll
    for (int i = 0; i < 8; ++i) { mrow[i] = -__builtin_inff(); lrow[i] = 0.0f; }

    const float scale = 1.0f / 4096.0f;   // reference divides by D_HEAD**2
    const int   nkv   = (qt + 1) * 2;     // causal: only kv tiles with kv0 <= q_max
    const int   kcB   = (lane < 16) ? 0 : 8;

    // ---- prologue: stage kv tile 0 into buffer 0 ----
    unsigned short vreg[16];
    stageK(0, 0);
    loadV(0, vreg);
    storeV(0, vreg);
    wait_async0();
    __syncthreads();

    for (int kt = 0; kt < nkv; ++kt) {
        const int  cur = kt & 1;
        const int  kv0 = kt * 64;
        const bool nxt = (kt + 1) < nkv;

        // prefetch next kv tile while computing this one
        if (nxt) {
            stageK(kv0 + 64, cur ^ 1);
            loadV(kv0 + 64, vreg);
        }

        // ---- scores: S[16 x 64] = Q @ K^T ----
        v8f sacc[4];
#pragma unroll
        for (int nt = 0; nt < 4; ++nt)
#pragma unroll
            for (int i = 0; i < 8; ++i) sacc[nt][i] = 0.0f;
#pragma unroll
        for (int ks = 0; ks < 2; ++ks)
#pragma unroll
            for (int nt = 0; nt < 4; ++nt) {
                FragBF kf;
                int n = nt * 16 + (lane & 15);
                kf.u[0] = *(const v4u*)&Ks[cur][n * 72 + ks * 32 + kcB];
                kf.u[1] = *(const v4u*)&Ks[cur][n * 72 + ks * 32 + kcB + 16];
                sacc[nt] = __builtin_amdgcn_wmma_f32_16x16x32_bf16(
                    false, qf[ks].bf, false, kf.bf, (short)0, sacc[nt], false, false);
            }

        // ---- scale + causal + padding mask, track row max ----
        float tmax[8];
#pragma unroll
        for (int i = 0; i < 8; ++i) tmax[i] = -__builtin_inff();
#pragma unroll
        for (int nt = 0; nt < 4; ++nt) {
            int col = kv0 + nt * 16 + (lane & 15);
            int pm  = mask[b * SEQ + col];
#pragma unroll
            for (int i = 0; i < 8; ++i) {
                int row = q0 + w * 16 + i + ((lane >= 16) ? 8 : 0);
                float s = sacc[nt][i] * scale;
                if (col > row || pm == 0) s = -__builtin_inff();
                sacc[nt][i] = s;
                tmax[i] = fmaxf(tmax[i], s);
            }
        }
        // half-wave (16-lane) butterfly reductions (wave32 C-layout rows)
#pragma unroll
        for (int i = 0; i < 8; ++i)
#pragma unroll
            for (int off = 1; off < 16; off <<= 1)
                tmax[i] = fmaxf(tmax[i], __shfl_xor(tmax[i], off, 32));

        // ---- online softmax update ----
        float mu[8], alpha[8], rsum[8];
#pragma unroll
        for (int i = 0; i < 8; ++i) {
            float mn = fmaxf(mrow[i], tmax[i]);
            mu[i]    = (mn == -__builtin_inff()) ? 0.0f : mn;
            alpha[i] = (mrow[i] == -__builtin_inff()) ? 0.0f : __expf(mrow[i] - mu[i]);
            mrow[i]  = mn;
            rsum[i]  = 0.0f;
        }
#pragma unroll
        for (int nt = 0; nt < 4; ++nt)
#pragma unroll
            for (int i = 0; i < 8; ++i) {
                float p = __expf(sacc[nt][i] - mu[i]);
                sacc[nt][i] = p;
                rsum[i] += p;
            }
#pragma unroll
        for (int i = 0; i < 8; ++i) {
#pragma unroll
            for (int off = 1; off < 16; off <<= 1) rsum[i] += __shfl_xor(rsum[i], off, 32);
            lrow[i] = lrow[i] * alpha[i] + rsum[i];
        }
#pragma unroll
        for (int nt = 0; nt < 4; ++nt)
#pragma unroll
            for (int i = 0; i < 8; ++i) oacc[nt][i] *= alpha[i];

        // ---- P (C-layout) -> LDS -> A-layout fragments (wave-private rows) ----
#pragma unroll
        for (int nt = 0; nt < 4; ++nt)
#pragma unroll
            for (int i = 0; i < 8; ++i) {
                int r = w * 16 + i + ((lane >= 16) ? 8 : 0);
                int c = nt * 16 + (lane & 15);
                QsPt[r * 72 + c] = f2bf(sacc[nt][i]);
            }
        asm volatile("s_wait_dscnt 0x0" ::: "memory");   // same-wave LDS RAW

        // ---- O += P @ V ----
#pragma unroll
        for (int ks = 0; ks < 2; ++ks) {
            FragBF pf;
            int r  = w * 16 + (lane & 15);
            int kc = (lane < 16) ? 0 : 8;
            pf.u[0] = *(const v4u*)&QsPt[r * 72 + ks * 32 + kc];
            pf.u[1] = *(const v4u*)&QsPt[r * 72 + ks * 32 + kc + 16];
#pragma unroll
            for (int nt = 0; nt < 4; ++nt) {
                FragBF vf;
                int n = nt * 16 + (lane & 15);
                vf.u[0] = *(const v4u*)&Vt[cur][n * 72 + ks * 32 + kcB];
                vf.u[1] = *(const v4u*)&Vt[cur][n * 72 + ks * 32 + kcB + 16];
                oacc[nt] = __builtin_amdgcn_wmma_f32_16x16x32_bf16(
                    false, pf.bf, false, vf.bf, (short)0, oacc[nt], false, false);
            }
        }

        // publish next tile: V transpose store + async K completion, one barrier
        if (nxt) storeV(cur ^ 1, vreg);
        wait_async0();
        __syncthreads();
    }

    // ---- normalize + write context [B,S,D] bf16 ----
#pragma unroll
    for (int nt = 0; nt < 4; ++nt)
#pragma unroll
        for (int i = 0; i < 8; ++i) {
            int row = q0 + w * 16 + i + ((lane >= 16) ? 8 : 0);
            int dh  = nt * 16 + (lane & 15);
            float v = oacc[nt][i] / lrow[i];
            Ctx[((size_t)b * SEQ + row) * D_MODEL + h * D_HEAD + dh] = f2bf(v);
        }
}

// ============================================================
extern "C" void kernel_launch(void* const* d_in, const int* in_sizes, int n_in,
                              void* d_out, int out_size, void* d_ws, size_t ws_size,
                              hipStream_t stream) {
    const float* query = (const float*)d_in[0];
    const float* key   = (const float*)d_in[1];
    const float* value = (const float*)d_in[2];
    const int*   mask  = (const int*)d_in[3];
    const float* Wq = (const float*)d_in[4];  const float* bq = (const float*)d_in[5];
    const float* Wk = (const float*)d_in[6];  const float* bk = (const float*)d_in[7];
    const float* Wv = (const float*)d_in[8];  const float* bv = (const float*)d_in[9];
    const float* Wo = (const float*)d_in[10]; const float* bo = (const float*)d_in[11];

    // scratch layout (bytes)
    char* ws = (char*)d_ws;
    const size_t WT_BYTES  = (size_t)D_MODEL * D_MODEL * 2;       // 2 MB each
    const size_t MAT_BYTES = (size_t)BATCH * SEQ * D_MODEL * 2;   // 16 MB each
    unsigned short* wtq = (unsigned short*)(ws + 0 * WT_BYTES);
    unsigned short* wtk = (unsigned short*)(ws + 1 * WT_BYTES);
    unsigned short* wtv = (unsigned short*)(ws + 2 * WT_BYTES);
    unsigned short* wto = (unsigned short*)(ws + 3 * WT_BYTES);
    unsigned short* qbf = (unsigned short*)(ws + 4 * WT_BYTES + 0 * MAT_BYTES);
    unsigned short* kbf = (unsigned short*)(ws + 4 * WT_BYTES + 1 * MAT_BYTES);
    unsigned short* vbf = (unsigned short*)(ws + 4 * WT_BYTES + 2 * MAT_BYTES);
    unsigned short* ctx = (unsigned short*)(ws + 4 * WT_BYTES + 3 * MAT_BYTES);

    // 1) weights -> bf16 transposed
    transpose_w<<<D_MODEL, 256, 0, stream>>>(Wq, wtq);
    transpose_w<<<D_MODEL, 256, 0, stream>>>(Wk, wtk);
    transpose_w<<<D_MODEL, 256, 0, stream>>>(Wv, wtv);
    transpose_w<<<D_MODEL, 256, 0, stream>>>(Wo, wto);

    // 2) QKV projections (f32 in -> bf16 [B,H,S,Dh] out)
    dim3 gg(64, 4);
    gemm_wmma<0, 0><<<gg, 256, 0, stream>>>((const void*)query, wtq, bq, (void*)qbf);
    gemm_wmma<0, 0><<<gg, 256, 0, stream>>>((const void*)key,   wtk, bk, (void*)kbf);
    gemm_wmma<0, 0><<<gg, 256, 0, stream>>>((const void*)value, wtv, bv, (void*)vbf);

    // 3) causal flash attention -> bf16 context [B,S,D]
    flash_attn<<<dim3(16, 64), 256, 0, stream>>>(qbf, kbf, vbf, mask, ctx);

    // 4) output projection (bf16 in -> f32 out)
    gemm_wmma<1, 1><<<gg, 256, 0, stream>>>((const void*)ctx, wto, bo, d_out);
}
